// CE_GNN_Model_19739669692531
// MI455X (gfx1250) — compile-verified, matrix operand order checked by reading, not run on hardware
//
#include <hip/hip_runtime.h>

#define N_NODES 4096
#define EMB     256
#define HEADS   8
#define HDIM    32
#define LSTM_H  128
#define NCLS    3
#define CDIV(a,b) (((a)+(b)-1)/(b))

typedef __attribute__((ext_vector_type(16))) __bf16 v16bf;
typedef __attribute__((ext_vector_type(8)))  float  v8f;

union Frag { v16bf v; unsigned u[8]; };

static __device__ __forceinline__ unsigned short f2bf(float f) {
    unsigned u = __float_as_uint(f);
    u += 0x7FFFu + ((u >> 16) & 1u);       // round-to-nearest-even
    return (unsigned short)(u >> 16);
}
static __device__ __forceinline__ float sigm(float x) { return 1.f / (1.f + __expf(-x)); }
static __device__ __forceinline__ float rmax16(float v) {
    #pragma unroll
    for (int m = 1; m < 16; m <<= 1) v = fmaxf(v, __shfl_xor(v, m, 16));
    return v;
}
static __device__ __forceinline__ float rsum16(float v) {
    #pragma unroll
    for (int m = 1; m < 16; m <<= 1) v += __shfl_xor(v, m, 16);
    return v;
}
static __device__ __forceinline__ v8f vzero8() {
    v8f z = {0.f,0.f,0.f,0.f,0.f,0.f,0.f,0.f};
    return z;
}
static __device__ __forceinline__ v8f wmma_bf16(const Frag& a, const Frag& b, v8f c) {
    return __builtin_amdgcn_wmma_f32_16x16x32_bf16(false, a.v, false, b.v,
                                                   (short)0, c, false, false);
}
// bf16 16x32 A/B fragment K-offset for element pair j (0..7)
static __device__ __forceinline__ constexpr int kpair(int j) {
    return ((j & 4) << 2) + ((j & 3) << 1);
}

// ---------------------------------------------------------------- conversions
__global__ void k_cvt_bf16(const float* __restrict__ in, unsigned short* __restrict__ out, int n) {
    int i = blockIdx.x * blockDim.x + threadIdx.x;
    if (i < n) out[i] = f2bf(in[i]);
}
// in [R,C] row-major fp32 -> out [C,R] row-major bf16 (transpose)
__global__ void k_cvt_bf16_T(const float* __restrict__ in, unsigned short* __restrict__ out, int R, int C) {
    int i = blockIdx.x * blockDim.x + threadIdx.x;
    if (i >= R * C) return;
    int r = i / C, c = i - r * C;
    out[c * R + r] = f2bf(in[i]);
}
__global__ void k_vec_add(const float* __restrict__ a, const float* __restrict__ b,
                          float* __restrict__ c, int n) {
    int i = blockIdx.x * blockDim.x + threadIdx.x;
    if (i < n) c[i] = a[i] + b[i];
}

// ------------------------------------------------------- WMMA GEMM: C = A*W^T
// A [M,K] bf16 row-major, W [N,K] bf16 row-major (torch weight layout).
// One wave computes a 32x32 tile: 2 A-frags x 2 B-frags -> 4 WMMAs per K-step,
// each loaded fragment used twice (2x arithmetic intensity vs 16x16 tiles).
template<bool HAS_BIAS, bool WF32, bool WBF16>
__global__ __launch_bounds__(256) void k_gemm_xwt(
    const unsigned short* __restrict__ A, const unsigned short* __restrict__ W,
    const float* __restrict__ bias, float* __restrict__ Cf, unsigned short* __restrict__ Cb,
    int K, int lda, int ldw, int ldc, int numTiles, int tilesN)
{
    int wave = (blockIdx.x * blockDim.x + threadIdx.x) >> 5;
    if (wave >= numTiles) return;
    int tn = wave % tilesN, tm = wave / tilesN;
    int m0 = tm << 5, n0 = tn << 5;
    int lane = threadIdx.x & 31;
    int r15 = lane & 15, hh = lane >> 4;
    const unsigned short* arow0 = A + (size_t)(m0 + r15) * lda + (hh << 3);
    const unsigned short* arow1 = arow0 + (size_t)16 * lda;
    const unsigned short* wrow0 = W + (size_t)(n0 + r15) * ldw + (hh << 3);
    const unsigned short* wrow1 = wrow0 + (size_t)16 * ldw;
    v8f acc00 = vzero8(), acc01 = vzero8(), acc10 = vzero8(), acc11 = vzero8();
    for (int kb = 0; kb < K; kb += 32) {
        Frag a0, a1, b0, b1;
        #pragma unroll
        for (int j = 0; j < 8; ++j) {
            int ko = kb + kpair(j);
            a0.u[j] = *(const unsigned*)(arow0 + ko);
            a1.u[j] = *(const unsigned*)(arow1 + ko);
            b0.u[j] = *(const unsigned*)(wrow0 + ko);
            b1.u[j] = *(const unsigned*)(wrow1 + ko);
        }
        acc00 = wmma_bf16(a0, b0, acc00);
        acc01 = wmma_bf16(a0, b1, acc01);
        acc10 = wmma_bf16(a1, b0, acc10);
        acc11 = wmma_bf16(a1, b1, acc11);
    }
    int c0 = n0 + r15, c1 = c0 + 16;
    float bv0 = HAS_BIAS ? bias[c0] : 0.f;
    float bv1 = HAS_BIAS ? bias[c1] : 0.f;
    #pragma unroll
    for (int r = 0; r < 8; ++r) {
        int row0 = m0 + r + (hh << 3);
        int row1 = row0 + 16;
        float v00 = acc00[r] + bv0, v01 = acc01[r] + bv1;
        float v10 = acc10[r] + bv0, v11 = acc11[r] + bv1;
        if (WF32) {
            Cf[(size_t)row0 * ldc + c0] = v00;
            Cf[(size_t)row0 * ldc + c1] = v01;
            Cf[(size_t)row1 * ldc + c0] = v10;
            Cf[(size_t)row1 * ldc + c1] = v11;
        }
        if (WBF16) {
            Cb[(size_t)row0 * ldc + c0] = f2bf(v00);
            Cb[(size_t)row0 * ldc + c1] = f2bf(v01);
            Cb[(size_t)row1 * ldc + c0] = f2bf(v10);
            Cb[(size_t)row1 * ldc + c1] = f2bf(v11);
        }
    }
}

// ----------------------------------------------- flash attention (wave-level)
// qkv bf16 [L, 768] (Q|K|V each 256 = 8 heads x 32). One wave = (head, 16 rows).
__global__ __launch_bounds__(256) void k_mha_flash(
    const unsigned short* __restrict__ qkv, unsigned short* __restrict__ obf, int L)
{
    __shared__ unsigned short pbuf[8][16 * 32];
    int wave  = (blockIdx.x * blockDim.x + threadIdx.x) >> 5;
    int wInB  = threadIdx.x >> 5;
    int tilesL = L >> 4;
    int h  = wave / tilesL;
    int l0 = (wave % tilesL) << 4;
    int lane = threadIdx.x & 31;
    int r15 = lane & 15, hh = lane >> 4;
    const int ldq = 3 * EMB;  // 768

    Frag qf;
    {
        const unsigned short* qrow = qkv + (size_t)(l0 + r15) * ldq + h * HDIM + (hh << 3);
        #pragma unroll
        for (int j = 0; j < 8; ++j) qf.u[j] = *(const unsigned*)(qrow + kpair(j));
    }
    v8f o0 = vzero8(), o1 = vzero8();
    float mrun[8], lrun[8];
    #pragma unroll
    for (int r = 0; r < 8; ++r) { mrun[r] = -1e30f; lrun[r] = 0.f; }
    const float scale = 0.17677669529663687f;  // 1/sqrt(32)

    for (int mb = 0; mb < L; mb += 32) {
        // K block: B-fragments, [keys, d] contiguous in d -> dword pair loads
        Frag kb0, kb1;
        const unsigned short* kbase = qkv + (size_t)(mb + r15) * ldq + EMB + h * HDIM + (hh << 3);
        #pragma unroll
        for (int j = 0; j < 8; ++j) {
            int ko = kpair(j);                       // immediate offsets
            kb0.u[j] = *(const unsigned*)(kbase + ko);
            kb1.u[j] = *(const unsigned*)(kbase + (size_t)16 * ldq + ko);
        }
        v8f s0 = wmma_bf16(qf, kb0, vzero8());
        v8f s1 = wmma_bf16(qf, kb1, vzero8());
        // online softmax per row (16-lane group reductions)
        #pragma unroll
        for (int r = 0; r < 8; ++r) {
            float a0 = s0[r] * scale, a1 = s1[r] * scale;
            float mx = rmax16(fmaxf(a0, a1));
            float mnew  = fmaxf(mrun[r], mx);
            float alpha = __expf(mrun[r] - mnew);
            float p0 = __expf(a0 - mnew), p1 = __expf(a1 - mnew);
            lrun[r] = lrun[r] * alpha + rsum16(p0 + p1);
            mrun[r] = mnew;
            o0[r] *= alpha; o1[r] *= alpha;
            int row = r + (hh << 3);
            pbuf[wInB][row * 32 + r15]      = f2bf(p0);
            pbuf[wInB][row * 32 + 16 + r15] = f2bf(p1);
        }
        asm volatile("s_wait_dscnt 0" ::: "memory");  // in-wave LDS transpose
        Frag pf;
        #pragma unroll
        for (int j = 0; j < 8; ++j)
            pf.u[j] = *(const unsigned*)(&pbuf[wInB][r15 * 32 + (hh << 3) + kpair(j)]);
        // V block: [K,N] layout; kpair(j)*ldq are compile-time immediate offsets
        Frag vb0, vb1;
        const unsigned short* vrow = qkv + (size_t)(mb + (hh << 3)) * ldq + 2 * EMB + h * HDIM + r15;
        #pragma unroll
        for (int j = 0; j < 8; ++j) {
            const unsigned short* p0 = vrow + (size_t)kpair(j) * ldq;
            unsigned lo0 = p0[0];
            unsigned hi0 = p0[ldq];
            vb0.u[j] = lo0 | (hi0 << 16);
            unsigned lo1 = p0[16];
            unsigned hi1 = p0[ldq + 16];
            vb1.u[j] = lo1 | (hi1 << 16);
        }
        o0 = wmma_bf16(pf, vb0, o0);
        o1 = wmma_bf16(pf, vb1, o1);
    }
    #pragma unroll
    for (int r = 0; r < 8; ++r) {
        int row = l0 + r + (hh << 3);
        float inv = 1.f / lrun[r];
        obf[(size_t)row * EMB + h * HDIM + r15]      = f2bf(o0[r] * inv);
        obf[(size_t)row * EMB + h * HDIM + 16 + r15] = f2bf(o1[r] * inv);
    }
}

// ---------------------------------------------------- sequential biLSTM scan
// G* [T, 512] = x@Wih^T + bih + bhh precomputed. Block 0: forward, 1: backward.
__global__ __launch_bounds__(128) void k_lstm_seq(
    const float* __restrict__ Gf, const float* __restrict__ Gb,
    const float* __restrict__ whhF, const float* __restrict__ whhB,
    float* __restrict__ ctx, int T)
{
    __shared__ float hsh[LSTM_H];
    int tid = threadIdx.x;
    bool rev = (blockIdx.x == 1);
    const float* G   = rev ? Gb   : Gf;
    const float* whh = rev ? whhB : whhF;
    int colOff = rev ? LSTM_H : 0;
    float c = 0.f;
    hsh[tid] = 0.f;
    __syncthreads();
    const float* wi = whh + (size_t)tid * LSTM_H;
    const float* wf = whh + (size_t)(LSTM_H + tid) * LSTM_H;
    const float* wg = whh + (size_t)(2 * LSTM_H + tid) * LSTM_H;
    const float* wo = whh + (size_t)(3 * LSTM_H + tid) * LSTM_H;
    for (int s = 0; s < T; ++s) {
        int t = rev ? (T - 1 - s) : s;
        const float* g = G + (size_t)t * 4 * LSTM_H;
        float gi = g[tid], gf2 = g[LSTM_H + tid], gg = g[2 * LSTM_H + tid], go = g[3 * LSTM_H + tid];
        #pragma unroll 4
        for (int k = 0; k < LSTM_H; ++k) {
            float hk = hsh[k];
            gi  = fmaf(wi[k], hk, gi);
            gf2 = fmaf(wf[k], hk, gf2);
            gg  = fmaf(wg[k], hk, gg);
            go  = fmaf(wo[k], hk, go);
        }
        c = sigm(gf2) * c + sigm(gi) * tanhf(gg);
        float hnew = sigm(go) * tanhf(c);
        __syncthreads();
        hsh[tid] = hnew;
        ctx[(size_t)t * EMB + colOff + tid] = hnew;
        __syncthreads();
    }
}

// --------------------------------------------------------------- GCN helpers
__global__ void k_fill1(float* __restrict__ p, int n) {
    int i = blockIdx.x * blockDim.x + threadIdx.x;
    if (i < n) p[i] = 1.f;
}
__global__ void k_deg_accum(const int* __restrict__ dst, float* __restrict__ deg, int E) {
    int i = blockIdx.x * blockDim.x + threadIdx.x;
    if (i < E) atomicAdd(&deg[dst[i]], 1.f);
}
__global__ void k_rsqrt(float* __restrict__ p, int n) {
    int i = blockIdx.x * blockDim.x + threadIdx.x;
    if (i < n) p[i] = rsqrtf(p[i]);
}
// acc = bias + dinv[n]^2 * xw[n,:]   (self-loop term + init)
__global__ void k_gcn_self_init(const float* __restrict__ xw, const float* __restrict__ dinv,
                                const float* __restrict__ bias, float* __restrict__ acc, int n) {
    int i = blockIdx.x * blockDim.x + threadIdx.x;
    if (i >= n) return;
    int node = i >> 8, c = i & 255;
    float di = dinv[node];
    acc[i] = bias[c] + di * di * xw[i];
}
// scatter: acc[dst] += dinv[src]*dinv[dst] * xw[src]; 64 threads/edge, float4 each
__global__ void k_gcn_scatter(const int* __restrict__ src, const int* __restrict__ dst,
                              const float* __restrict__ xw, const float* __restrict__ dinv,
                              float* __restrict__ acc, int E) {
    long long i = (long long)blockIdx.x * blockDim.x + threadIdx.x;
    if (i >= (long long)E * 64) return;
    int e = (int)(i >> 6);
    int c = ((int)i & 63) << 2;
    int s = src[e], d = dst[e];
    float w = dinv[s] * dinv[d];
    const float4 v = *(const float4*)(xw + (size_t)s * EMB + c);
    float* a = acc + (size_t)d * EMB + c;
    atomicAdd(a + 0, w * v.x);
    atomicAdd(a + 1, w * v.y);
    atomicAdd(a + 2, w * v.z);
    atomicAdd(a + 3, w * v.w);
}
__global__ void k_relu_bf(const float* __restrict__ acc, float* __restrict__ g,
                          unsigned short* __restrict__ gbf, int n) {
    int i = blockIdx.x * blockDim.x + threadIdx.x;
    if (i >= n) return;
    float v = fmaxf(acc[i], 0.f);
    g[i] = v;
    gbf[i] = f2bf(v);
}
// logits = g @ fcW^T + fcB; softmax over 3 classes
__global__ void k_fc_softmax(const float* __restrict__ g, const float* __restrict__ fw,
                             const float* __restrict__ fb, float* __restrict__ out, int N) {
    int n = blockIdx.x * blockDim.x + threadIdx.x;
    if (n >= N) return;
    float l[NCLS];
    #pragma unroll
    for (int c = 0; c < NCLS; ++c) {
        float a = fb[c];
        const float* gr = g + (size_t)n * EMB;
        const float* wr = fw + (size_t)c * EMB;
        for (int k = 0; k < EMB; ++k) a = fmaf(gr[k], wr[k], a);
        l[c] = a;
    }
    float m = fmaxf(l[0], fmaxf(l[1], l[2]));
    float e0 = __expf(l[0] - m), e1 = __expf(l[1] - m), e2 = __expf(l[2] - m);
    float inv = 1.f / (e0 + e1 + e2);
    out[n * 3 + 0] = e0 * inv;
    out[n * 3 + 1] = e1 * inv;
    out[n * 3 + 2] = e2 * inv;
}

// ------------------------------------------------------------------- driver
extern "C" void kernel_launch(void* const* d_in, const int* in_sizes, int n_in,
                              void* d_out, int out_size, void* d_ws, size_t ws_size,
                              hipStream_t stream) {
    const float* x    = (const float*)d_in[0];
    const int*   eidx = (const int*)d_in[1];
    const float* inW  = (const float*)d_in[2];
    const float* inB  = (const float*)d_in[3];
    const float* outW = (const float*)d_in[4];
    const float* outB = (const float*)d_in[5];
    const float* wihF = (const float*)d_in[6];
    const float* whhF = (const float*)d_in[7];
    const float* bihF = (const float*)d_in[8];
    const float* bhhF = (const float*)d_in[9];
    const float* wihB = (const float*)d_in[10];
    const float* whhB = (const float*)d_in[11];
    const float* bihB = (const float*)d_in[12];
    const float* bhhB = (const float*)d_in[13];
    const float* gcnW = (const float*)d_in[14];
    const float* gcnB = (const float*)d_in[15];
    const float* fcW  = (const float*)d_in[16];
    const float* fcB  = (const float*)d_in[17];
    float* out = (float*)d_out;

    const int L = N_NODES;
    const int E = in_sizes[1] / 2;

    char* base = (char*)d_ws;
    size_t off = 0;
    auto alloc = [&](size_t b) -> void* {
        void* p = base + off;
        off = (off + b + 255) & ~(size_t)255;
        return p;
    };
    unsigned short* xbf    = (unsigned short*)alloc((size_t)L * EMB * 2);
    unsigned short* winbf  = (unsigned short*)alloc((size_t)768 * EMB * 2);
    unsigned short* woutbf = (unsigned short*)alloc((size_t)EMB * EMB * 2);
    unsigned short* wihfbf = (unsigned short*)alloc((size_t)512 * EMB * 2);
    unsigned short* wihbbf = (unsigned short*)alloc((size_t)512 * EMB * 2);
    unsigned short* gcnwbf = (unsigned short*)alloc((size_t)3 * EMB * EMB * 2);
    unsigned short* qkvbf  = (unsigned short*)alloc((size_t)L * 768 * 2);
    unsigned short* attnbf = (unsigned short*)alloc((size_t)L * EMB * 2);
    unsigned short* abf    = (unsigned short*)alloc((size_t)L * EMB * 2);
    float* biasF = (float*)alloc(512 * 4);
    float* biasB = (float*)alloc(512 * 4);
    float* Gf    = (float*)alloc((size_t)L * 512 * 4);
    float* Gb    = (float*)alloc((size_t)L * 512 * 4);
    float* ctx   = (float*)alloc((size_t)L * EMB * 4);
    unsigned short* ctxbf = (unsigned short*)alloc((size_t)L * EMB * 2);
    float* dinv  = (float*)alloc((size_t)L * 4);
    float* xw    = (float*)alloc((size_t)L * EMB * 4);
    float* acc   = (float*)alloc((size_t)L * EMB * 4);
    float* gout  = (float*)alloc((size_t)L * EMB * 4);
    unsigned short* gbf = (unsigned short*)alloc((size_t)L * EMB * 2);

    // ---- precision conversions
    k_cvt_bf16<<<CDIV(L * EMB, 256), 256, 0, stream>>>(x, xbf, L * EMB);
    k_cvt_bf16<<<CDIV(768 * EMB, 256), 256, 0, stream>>>(inW, winbf, 768 * EMB);
    k_cvt_bf16<<<CDIV(EMB * EMB, 256), 256, 0, stream>>>(outW, woutbf, EMB * EMB);
    k_cvt_bf16<<<CDIV(512 * EMB, 256), 256, 0, stream>>>(wihF, wihfbf, 512 * EMB);
    k_cvt_bf16<<<CDIV(512 * EMB, 256), 256, 0, stream>>>(wihB, wihbbf, 512 * EMB);
    for (int l = 0; l < 3; ++l)
        k_cvt_bf16_T<<<CDIV(EMB * EMB, 256), 256, 0, stream>>>(
            gcnW + (size_t)l * EMB * EMB, gcnwbf + (size_t)l * EMB * EMB, EMB, EMB);
    k_vec_add<<<2, 256, 0, stream>>>(bihF, bhhF, biasF, 512);
    k_vec_add<<<2, 256, 0, stream>>>(bihB, bhhB, biasB, 512);

    // ---- attention: QKV projection -> flash -> out projection
    {
        int tiles = (L / 32) * (768 / 32);
        k_gemm_xwt<true, false, true><<<CDIV(tiles, 8), 256, 0, stream>>>(
            xbf, winbf, inB, nullptr, qkvbf, EMB, EMB, EMB, 768, tiles, 768 / 32);
    }
    k_mha_flash<<<(HEADS * (L / 16)) / 8, 256, 0, stream>>>(qkvbf, attnbf, L);
    {
        int tiles = (L / 32) * (EMB / 32);
        k_gemm_xwt<true, false, true><<<CDIV(tiles, 8), 256, 0, stream>>>(
            attnbf, woutbf, outB, nullptr, abf, EMB, EMB, EMB, EMB, tiles, EMB / 32);
    }

    // ---- LSTM: precompute input gates (WMMA), then sequential scan
    {
        int tiles = (L / 32) * (512 / 32);
        k_gemm_xwt<true, true, false><<<CDIV(tiles, 8), 256, 0, stream>>>(
            abf, wihfbf, biasF, Gf, nullptr, EMB, EMB, EMB, 512, tiles, 512 / 32);
        k_gemm_xwt<true, true, false><<<CDIV(tiles, 8), 256, 0, stream>>>(
            abf, wihbbf, biasB, Gb, nullptr, EMB, EMB, EMB, 512, tiles, 512 / 32);
    }
    k_lstm_seq<<<2, 128, 0, stream>>>(Gf, Gb, whhF, whhB, ctx, L);
    k_cvt_bf16<<<CDIV(L * EMB, 256), 256, 0, stream>>>(ctx, ctxbf, L * EMB);

    // ---- GCN: degree norm, 3 layers of (WMMA GEMM, scatter, ReLU)
    k_fill1<<<CDIV(L, 256), 256, 0, stream>>>(dinv, L);
    k_deg_accum<<<CDIV(E, 256), 256, 0, stream>>>(eidx + E, dinv, E);
    k_rsqrt<<<CDIV(L, 256), 256, 0, stream>>>(dinv, L);
    for (int l = 0; l < 3; ++l) {
        const unsigned short* cur = (l == 0) ? ctxbf : gbf;
        int tiles = (L / 32) * (EMB / 32);
        k_gemm_xwt<false, true, false><<<CDIV(tiles, 8), 256, 0, stream>>>(
            cur, gcnwbf + (size_t)l * EMB * EMB, nullptr, xw, nullptr,
            EMB, EMB, EMB, EMB, tiles, EMB / 32);
        k_gcn_self_init<<<CDIV(L * EMB, 256), 256, 0, stream>>>(
            xw, dinv, gcnB + (size_t)l * EMB, acc, L * EMB);
        long long tot = (long long)E * 64;
        k_gcn_scatter<<<(unsigned)CDIV(tot, 256), 256, 0, stream>>>(
            eidx, eidx + E, xw, dinv, acc, E);
        k_relu_bf<<<CDIV(L * EMB, 256), 256, 0, stream>>>(acc, gout, gbf, L * EMB);
    }

    // ---- classifier + softmax
    k_fc_softmax<<<CDIV(L, 256), 256, 0, stream>>>(gout, fcW, fcB, out, L);
}
